// Attention_39058432589847
// MI455X (gfx1250) — compile-verified, matrix operand order checked by reading, not run on hardware
//
#include <hip/hip_runtime.h>

#define E_DIM   1024
#define NHEADS  16
#define HDIM    64
#define BATCH   4
#define SEQ     2048

typedef __bf16 bf16_t;
typedef __attribute__((ext_vector_type(16))) __bf16 v16bf;
typedef __attribute__((ext_vector_type(8)))  __bf16 bf8;
typedef __attribute__((ext_vector_type(8)))  float  v8f;

__device__ __forceinline__ bf16_t f2bf(float f) {
    unsigned u = __builtin_bit_cast(unsigned, f);
    unsigned r = (u + 0x7FFFu + ((u >> 16) & 1u)) >> 16;
    unsigned short s = (unsigned short)r;
    return __builtin_bit_cast(bf16_t, s);
}

union Frag16 { v16bf v; bf8 h[2]; };

// Async global->LDS copy of 32 bytes (two b128 halves sharing one address
// pair; INST_OFFSET is added to both the LDS and global addresses per the
// CDNA5 ISA).  Tracked on ASYNCcnt; pair with async_wait() before the
// consuming barrier.  Low 32 bits of a generic LDS pointer are the DS offset.
__device__ __forceinline__ void async_copy32(void* lds_dst, const void* gsrc) {
    unsigned ldsa = (unsigned)(uintptr_t)lds_dst;
    asm volatile("global_load_async_to_lds_b128 %0, %1, off offset:0"
                 :: "v"(ldsa), "v"(gsrc) : "memory");
    asm volatile("global_load_async_to_lds_b128 %0, %1, off offset:16"
                 :: "v"(ldsa), "v"(gsrc) : "memory");
}

__device__ __forceinline__ void async_wait() {
    asm volatile("s_wait_asynccnt 0x0" ::: "memory");
}

// Load a 16xK32 WMMA fragment (A or B role, identical lane pattern) from a
// row-major bf16 tile with row stride `stride` elements, K offset `kofs`.
// Lane layout per CDNA5 ISA: lanes 0-15 hold K 0-7 & 16-23, lanes 16-31 hold
// K 8-15 & 24-31, row/col = lane & 15.  Two 16B ds_load_b128 per lane.
__device__ __forceinline__ v16bf load_frag(const bf16_t* base, int stride, int kofs) {
    const int lane = threadIdx.x & 31;
    const int lh   = lane >> 4;
    const int r    = lane & 15;
    const bf16_t* p = base + r * stride + kofs + lh * 8;
    Frag16 f;
    f.h[0] = *(const bf8*)(p);
    f.h[1] = *(const bf8*)(p + 16);
    return f.v;
}

__device__ __forceinline__ bf8 cvt8(float4 a, float4 b) {
    bf8 r;
    r[0] = f2bf(a.x); r[1] = f2bf(a.y); r[2] = f2bf(a.z); r[3] = f2bf(a.w);
    r[4] = f2bf(b.x); r[5] = f2bf(b.y); r[6] = f2bf(b.z); r[7] = f2bf(b.w);
    return r;
}

// C[M,N] = A[M,K] @ W[N,K]^T + bias.  A is f32 (ABF=false) or bf16 (ABF=true),
// W/bias always f32, output bf16 (OBF=true) or f32.  BM=BN=128, BK=32,
// 8 waves -> 4x2 wave grid, each wave owns a 32x64 tile = 2x4 WMMA tiles.
template<bool ABF, bool OBF>
__global__ __launch_bounds__(256) void gemm_bias_kernel(
    const void* __restrict__ Av, const float* __restrict__ Wf,
    const float* __restrict__ bias, void* __restrict__ Ov,
    int M, int N, int K)
{
    constexpr int BM = 128, BN = 128, BK = 32, LDT = BK + 8;  // 80B rows, 16B aligned
    __shared__ bf16_t As[BM][LDT];
    __shared__ bf16_t Bs[BN][LDT];

    const int tid  = threadIdx.x;
    const int wave = tid >> 5;
    const int lane = tid & 31;
    const int lh   = lane >> 4;
    const int ln   = lane & 15;
    const int wm   = wave & 3;    // wave row (M)
    const int wn   = wave >> 2;   // wave col (N)
    const int blockM = blockIdx.y * BM;
    const int blockN = blockIdx.x * BN;

    v8f zero = {};
    v8f acc[2][4];
    for (int i = 0; i < 2; ++i)
        for (int j = 0; j < 4; ++j) acc[i][j] = zero;

    const int srow = tid >> 1;        // 0..127
    const int scol = (tid & 1) * 16;  // 0 or 16

    for (int k0 = 0; k0 < K; k0 += BK) {
        // ---- stage A tile ----
        if (ABF) {
            // raw bf16 copy: use the CDNA5 async global->LDS path
            const bf16_t* A = (const bf16_t*)Av;
            async_copy32(&As[srow][scol],
                         A + (size_t)(blockM + srow) * K + k0 + scol);
        } else {
            // f32 -> bf16 conversion in flight (must go through VALU)
            const float* A = (const float*)Av;
            const float4* p = (const float4*)(A + (size_t)(blockM + srow) * K + k0 + scol);
            float4 x0 = p[0], x1 = p[1], x2 = p[2], x3 = p[3];
            *(bf8*)&As[srow][scol]     = cvt8(x0, x1);
            *(bf8*)&As[srow][scol + 8] = cvt8(x2, x3);
        }

        // ---- stage B tile from W (always f32 [N,K]) ----
        {
            const float4* p = (const float4*)(Wf + (size_t)(blockN + srow) * K + k0 + scol);
            float4 x0 = p[0], x1 = p[1], x2 = p[2], x3 = p[3];
            *(bf8*)&Bs[srow][scol]     = cvt8(x0, x1);
            *(bf8*)&Bs[srow][scol + 8] = cvt8(x2, x3);
        }

        // prefetch next K-step tiles into cache while this one computes
        if (k0 + BK < K) {
            if (!ABF)
                __builtin_prefetch((const float*)Av + (size_t)(blockM + srow) * K + k0 + BK + scol, 0, 1);
            __builtin_prefetch(Wf + (size_t)(blockN + srow) * K + k0 + BK + scol, 0, 1);
        }

        if (ABF) async_wait();
        __syncthreads();

        v16bf fa0 = load_frag(&As[wm * 32][0],      LDT, 0);
        v16bf fa1 = load_frag(&As[wm * 32 + 16][0], LDT, 0);
        v16bf fb0 = load_frag(&Bs[wn * 64][0],      LDT, 0);
        v16bf fb1 = load_frag(&Bs[wn * 64 + 16][0], LDT, 0);
        v16bf fb2 = load_frag(&Bs[wn * 64 + 32][0], LDT, 0);
        v16bf fb3 = load_frag(&Bs[wn * 64 + 48][0], LDT, 0);

        acc[0][0] = __builtin_amdgcn_wmma_f32_16x16x32_bf16(false, fa0, false, fb0, (short)0, acc[0][0], false, false);
        acc[0][1] = __builtin_amdgcn_wmma_f32_16x16x32_bf16(false, fa0, false, fb1, (short)0, acc[0][1], false, false);
        acc[0][2] = __builtin_amdgcn_wmma_f32_16x16x32_bf16(false, fa0, false, fb2, (short)0, acc[0][2], false, false);
        acc[0][3] = __builtin_amdgcn_wmma_f32_16x16x32_bf16(false, fa0, false, fb3, (short)0, acc[0][3], false, false);
        acc[1][0] = __builtin_amdgcn_wmma_f32_16x16x32_bf16(false, fa1, false, fb0, (short)0, acc[1][0], false, false);
        acc[1][1] = __builtin_amdgcn_wmma_f32_16x16x32_bf16(false, fa1, false, fb1, (short)0, acc[1][1], false, false);
        acc[1][2] = __builtin_amdgcn_wmma_f32_16x16x32_bf16(false, fa1, false, fb2, (short)0, acc[1][2], false, false);
        acc[1][3] = __builtin_amdgcn_wmma_f32_16x16x32_bf16(false, fa1, false, fb3, (short)0, acc[1][3], false, false);
        __syncthreads();
    }

    // ---- epilogue: C-layout row = e + 8*lh, col = ln ----
    for (int i = 0; i < 2; ++i) {
        const int rowbase = blockM + wm * 32 + i * 16 + lh * 8;
        for (int j = 0; j < 4; ++j) {
            const int col = blockN + wn * 64 + j * 16 + ln;
            const float bv = bias[col];
            for (int e = 0; e < 8; ++e) {
                const int row = rowbase + e;
                const float v = acc[i][j][e] + bv;
                if (OBF) ((bf16_t*)Ov)[(size_t)row * N + col] = f2bf(v);
                else     ((float*) Ov)[(size_t)row * N + col] = v;
            }
        }
    }
}

// Flash attention, causal.  Block = 128 q-rows of one (b,h); each wave owns 16
// rows.  KV tiles of 64 keys staged in LDS (K via async global->LDS DMA, V
// transposed through VGPRs), online softmax with wave32 shuffles, P
// round-trips through per-wave LDS to become A-fragments.
__global__ __launch_bounds__(256) void attn_kernel(
    const bf16_t* __restrict__ Qp, const bf16_t* __restrict__ Kp,
    const bf16_t* __restrict__ Vp, bf16_t* __restrict__ Op)
{
    constexpr int TQ = 128, TK = 64, LDK = 72;   // 144B rows, 16B aligned
    __shared__ bf16_t Ks[TK][LDK];       // [key][d]
    __shared__ bf16_t Vt[HDIM][LDK];     // [d][key]   (transposed V)
    __shared__ bf16_t Ps[8][16][LDK];    // per-wave P tile [qrow][key]

    const int tid  = threadIdx.x;
    const int wave = tid >> 5;
    const int lane = tid & 31;
    const int lh   = lane >> 4;
    const int ln   = lane & 15;

    const int q0 = blockIdx.x * TQ;
    const int h  = blockIdx.y;
    const int b  = blockIdx.z;

    const bf16_t* Qb = Qp + (size_t)b * SEQ * E_DIM + h * HDIM;
    const bf16_t* Kb = Kp + (size_t)b * SEQ * E_DIM + h * HDIM;
    const bf16_t* Vb = Vp + (size_t)b * SEQ * E_DIM + h * HDIM;
    bf16_t*       Ob = Op + (size_t)b * SEQ * E_DIM + h * HDIM;

    const int qr = q0 + wave * 16;

    // Q fragments (A-layout), loaded once from global
    v16bf aq[2];
    for (int c = 0; c < 2; ++c) {
        const bf16_t* p = Qb + (size_t)(qr + ln) * E_DIM + c * 32 + lh * 8;
        Frag16 f;
        f.h[0] = *(const bf8*)(p);
        f.h[1] = *(const bf8*)(p + 16);
        aq[c] = f.v;
    }

    v8f zero = {};
    v8f o_acc[4];
    float m_i[8], l_i[8];
    for (int e = 0; e < 8; ++e) { m_i[e] = -1e30f; l_i[e] = 0.f; }
    for (int j = 0; j < 4; ++j) o_acc[j] = zero;

    const int nt   = q0 / TK + 2;       // only tiles with k_start <= q0+127 (causal)
    const int srow = tid >> 2;          // 0..63
    const int sd   = (tid & 3) * 16;    // 0,16,32,48
    const float LOG2E = 1.44269504088896f;

    for (int kt = 0; kt < nt; ++kt) {
        const int k_start = kt * TK;

        // stage K tile via async global->LDS DMA (raw bf16 bytes)
        async_copy32(&Ks[srow][sd],
                     Kb + (size_t)(k_start + srow) * E_DIM + sd);

        // stage V tile transposed (needs VGPR shuffle, stays on VALU path)
        {
            const bf8* p = (const bf8*)(Vb + (size_t)(k_start + srow) * E_DIM + sd);
            bf8 v0 = p[0], v1 = p[1];
            for (int i = 0; i < 8; ++i) Vt[sd + i][srow]     = v0[i];
            for (int i = 0; i < 8; ++i) Vt[sd + 8 + i][srow] = v1[i];
        }
        async_wait();
        __syncthreads();

        // S = Q @ K^T  (16 x 64 per wave)
        v8f s[4];
        for (int j = 0; j < 4; ++j) s[j] = zero;
        for (int c = 0; c < 2; ++c) {
            for (int j = 0; j < 4; ++j) {
                v16bf fb = load_frag(&Ks[j * 16][0], LDK, c * 32);
                s[j] = __builtin_amdgcn_wmma_f32_16x16x32_bf16(false, aq[c], false, fb, (short)0, s[j], false, false);
            }
        }

        // scale + causal mask
        for (int j = 0; j < 4; ++j) {
            const int kg = k_start + j * 16 + ln;
            for (int e = 0; e < 8; ++e) {
                const int qg = qr + e + 8 * lh;
                const float v = s[j][e] * 0.125f;          // 1/sqrt(64)
                s[j][e] = (kg > qg) ? -1e30f : v;
            }
        }

        // online softmax: rows live in 16-lane groups (lh), one row per elem e
        float mnew[8], sc[8], rs[8];
        for (int e = 0; e < 8; ++e) {
            float lm = fmaxf(fmaxf(s[0][e], s[1][e]), fmaxf(s[2][e], s[3][e]));
            for (int msk = 8; msk >= 1; msk >>= 1)
                lm = fmaxf(lm, __shfl_xor(lm, msk, 32));
            mnew[e] = fmaxf(m_i[e], lm);
            sc[e]   = __builtin_amdgcn_exp2f((m_i[e] - mnew[e]) * LOG2E);
            rs[e]   = 0.f;
        }
        for (int j = 0; j < 4; ++j) {
            for (int e = 0; e < 8; ++e) {
                const float p = __builtin_amdgcn_exp2f((s[j][e] - mnew[e]) * LOG2E);
                s[j][e] = p;
                rs[e] += p;
            }
        }
        for (int e = 0; e < 8; ++e) {
            for (int msk = 8; msk >= 1; msk >>= 1)
                rs[e] += __shfl_xor(rs[e], msk, 32);
            l_i[e] = l_i[e] * sc[e] + rs[e];
            m_i[e] = mnew[e];
        }
        for (int j = 0; j < 4; ++j)
            for (int e = 0; e < 8; ++e)
                o_acc[j][e] = o_acc[j][e] * sc[e];

        // P (C-layout) -> LDS so it can be re-read as an A-fragment
        for (int j = 0; j < 4; ++j)
            for (int e = 0; e < 8; ++e)
                Ps[wave][e + 8 * lh][j * 16 + ln] = f2bf(s[j][e]);
        __syncthreads();

        // O += P @ V
        for (int c = 0; c < 2; ++c) {
            v16bf ap = load_frag(&Ps[wave][0][0], LDK, c * 32);
            for (int jd = 0; jd < 4; ++jd) {
                v16bf fv = load_frag(&Vt[jd * 16][0], LDK, c * 32);
                o_acc[jd] = __builtin_amdgcn_wmma_f32_16x16x32_bf16(false, ap, false, fv, (short)0, o_acc[jd], false, false);
            }
        }
        __syncthreads();
    }

    // epilogue: O /= l, store bf16 into [B,S,E] at head column block
    for (int jd = 0; jd < 4; ++jd) {
        const int col = jd * 16 + ln;
        for (int e = 0; e < 8; ++e) {
            const int row = qr + e + 8 * lh;
            const float v = o_acc[jd][e] / l_i[e];
            Ob[(size_t)row * E_DIM + col] = f2bf(v);
        }
    }
}

extern "C" void kernel_launch(void* const* d_in, const int* in_sizes, int n_in,
                              void* d_out, int out_size, void* d_ws, size_t ws_size,
                              hipStream_t stream)
{
    (void)in_sizes; (void)n_in; (void)out_size; (void)ws_size;

    const float* query = (const float*)d_in[0];
    const float* key   = (const float*)d_in[1];
    const float* value = (const float*)d_in[2];
    const float* Wq    = (const float*)d_in[3];
    const float* bq    = (const float*)d_in[4];
    const float* Wk    = (const float*)d_in[5];
    const float* bk    = (const float*)d_in[6];
    const float* Wv    = (const float*)d_in[7];
    const float* bv    = (const float*)d_in[8];
    const float* Wo    = (const float*)d_in[9];
    const float* bo    = (const float*)d_in[10];
    float* out = (float*)d_out;

    const int M = BATCH * SEQ;   // 8192
    const int N = E_DIM, K = E_DIM;

    bf16_t* Qp = (bf16_t*)d_ws;                 // [8192,1024] bf16
    bf16_t* Kp = Qp + (size_t)M * E_DIM;
    bf16_t* Vp = Kp + (size_t)M * E_DIM;
    bf16_t* Ap = Vp + (size_t)M * E_DIM;        // attention output, bf16

    dim3 bb(256);
    dim3 gg(N / 128, M / 128);
    gemm_bias_kernel<false, true><<<gg, bb, 0, stream>>>(query, Wq, bq, Qp, M, N, K);
    gemm_bias_kernel<false, true><<<gg, bb, 0, stream>>>(key,   Wk, bk, Kp, M, N, K);
    gemm_bias_kernel<false, true><<<gg, bb, 0, stream>>>(value, Wv, bv, Vp, M, N, K);

    dim3 ga(SEQ / 128, NHEADS, BATCH);
    attn_kernel<<<ga, bb, 0, stream>>>(Qp, Kp, Vp, Ap);

    gemm_bias_kernel<true, false><<<gg, bb, 0, stream>>>(Ap, Wo, bo, out, M, N, K);
}
